// GATLayer_17085379903855
// MI455X (gfx1250) — matrix operand order repaired and checked by reading, not compiled, not run on hardware
//
#include <hip/hip_runtime.h>
#include <stdint.h>

#define B_DIM 8
#define N_DIM 2048
#define F_DIM 128
#define NEG_BIG_F (-9000000000000000.0f)
#define SLOPE 0.2f

typedef __attribute__((ext_vector_type(16))) __bf16 v16bf;
typedef __attribute__((ext_vector_type(8)))  float  v8f;
typedef __attribute__((ext_vector_type(2)))  float  v2f;
typedef __attribute__((ext_vector_type(8)))  unsigned int v8u;
typedef __attribute__((ext_vector_type(4)))  unsigned int v4u;
typedef __attribute__((ext_vector_type(8)))  int v8i;
typedef __attribute__((ext_vector_type(4)))  int v4i;

#if __has_builtin(__builtin_amdgcn_tensor_load_to_lds) && \
    __has_builtin(__builtin_amdgcn_s_wait_tensorcnt)
#define HAVE_TDM 1
#endif

__device__ __forceinline__ unsigned f2bf_bits(float f) {
  unsigned u = __builtin_bit_cast(unsigned, f);
  return (u + 0x7FFFu + ((u >> 16) & 1u)) >> 16;   // round-to-nearest-even
}

#ifdef HAVE_TDM
// Issue one Tensor-Data-Mover load: 2D tile (or 1D if tile_d1==0) of 4-byte
// elements, global -> LDS, optional LDS padding (pad_interval encoded per D#:
// 5 -> every 64 dwords, pad_amount field 0 -> 1 dword).
// 6-arg builtin form (clang-23 / therock headers): (g0, g1, g2, g3, g4, cpol).
__device__ __forceinline__ void tdm_load_4B(unsigned lds_off, const void* gaddr,
                                            unsigned tensor_d0, unsigned tensor_d1,
                                            unsigned tile_d0, unsigned tile_d1,
                                            unsigned stride0,
                                            unsigned pad_en, unsigned pad_interval) {
  const unsigned long long ga = (unsigned long long)gaddr;
  v4u g0;
  g0[0] = 1u;                                   // count=1, user mode, no gather
  g0[1] = lds_off;                              // lds_addr (bytes)
  g0[2] = (unsigned)ga;                         // global_addr[31:0]
  g0[3] = (unsigned)(ga >> 32) | (2u << 30);    // global_addr[56:32] | type=2
  v8i g1;
  g1[0] = (int)((2u << 16) | (pad_en << 20) | (pad_interval << 22)); // data_size=4B
  g1[1] = (int)((tensor_d0 & 0xFFFFu) << 16);                        // tensor_dim0 lo
  g1[2] = (int)(((tensor_d0 >> 16) & 0xFFFFu) | ((tensor_d1 & 0xFFFFu) << 16));
  g1[3] = (int)(((tensor_d1 >> 16) & 0xFFFFu) | ((tile_d0 & 0xFFFFu) << 16));
  g1[4] = (int)(tile_d1 & 0xFFFFu);                                  // tile_dim1 (0=1D)
  g1[5] = (int)stride0;                                              // dim0_stride lo
  g1[6] = 0;
  g1[7] = 0;
  const v4i g2 = {0, 0, 0, 0};
  const v4i g3 = {0, 0, 0, 0};
  const v8i g4 = {0, 0, 0, 0, 0, 0, 0, 0};
  __builtin_amdgcn_tensor_load_to_lds(g0, g1, g2, g3, g4, 0);
}
#endif

// ---------------------------------------------------------------------------
// Kernel 1: pack adj>0 into bitmask (2048 x 64 words). 32x less mask traffic.
// ---------------------------------------------------------------------------
__global__ __launch_bounds__(256) void k_pack(const int* __restrict__ adj,
                                              unsigned* __restrict__ adjbits) {
  const int idx = blockIdx.x * 256 + threadIdx.x;     // row*64 + word
  const int row = idx >> 6, w = idx & 63;
  const int* p = adj + (size_t)row * N_DIM + w * 32;
  unsigned bits = 0;
  #pragma unroll
  for (int j = 0; j < 32; ++j) bits |= (p[j] > 0 ? 1u : 0u) << j;
  adjbits[idx] = bits;
}

// ---------------------------------------------------------------------------
// Kernel 2: Wh = h @ W in exact f32 via v_wmma_f32_16x16x4_f32.
// Grid: B * N/16 blocks of 256 (8 waves). Wave w computes 16-col block f0=16w.
// ---------------------------------------------------------------------------
__global__ __launch_bounds__(256) void k_wh(const float* __restrict__ h,
                                            const float* __restrict__ W,
                                            float* __restrict__ Wh) {
  __shared__ __align__(16) float sh[16 * F_DIM];
  const int blk  = blockIdx.x;
  const int b    = blk >> 7;             // 128 row-tiles per batch
  const int i0   = (blk & 127) << 4;
  const int t    = threadIdx.x;
  const int lane = t & 31;
  const int wave = t >> 5;
  const int M    = lane & 15;
  const int half = lane >> 4;
  const int f0   = wave * 16;

  const float* hsrc = h + ((size_t)b * N_DIM + i0) * F_DIM;
  for (int idx = t; idx < 16 * F_DIM / 4; idx += 256)
    ((float4*)sh)[idx] = ((const float4*)hsrc)[idx];
  __syncthreads();

  v8f acc = {};
#if __has_builtin(__builtin_amdgcn_wmma_f32_16x16x4_f32)
  // A 16x4 f32: lane=M, VGPR r -> K = r + 2*half.  B 4x16: K = r + 2*half, N = M.
  const float* wp = W + f0 + M;
  #pragma unroll
  for (int k = 0; k < F_DIM; k += 4) {
    const float2 av = *(const float2*)&sh[M * F_DIM + k + (half << 1)];
    v2f a;  a[0] = av.x; a[1] = av.y;
    v2f bm;
    bm[0] = wp[(size_t)(k + (half << 1) + 0) * F_DIM];
    bm[1] = wp[(size_t)(k + (half << 1) + 1) * F_DIM];
    acc = __builtin_amdgcn_wmma_f32_16x16x4_f32(false, a, false, bm,
                                                (short)0, acc, false, false);
  }
#else
  // Scalar fallback with identical D layout.
  for (int k = 0; k < F_DIM; ++k) {
    const float wv = W[(size_t)k * F_DIM + f0 + M];
    #pragma unroll
    for (int r = 0; r < 8; ++r)
      acc[r] += sh[(r + (half << 3)) * F_DIM + k] * wv;
  }
#endif
  float* op = Wh + ((size_t)b * N_DIM + i0) * F_DIM + f0 + M;
  #pragma unroll
  for (int r = 0; r < 8; ++r)
    op[(size_t)(r + (half << 3)) * F_DIM] = acc[r];
}

// ---------------------------------------------------------------------------
// Kernel 3: per-node scores s_src/s_dst + bf16 copy of Wh for the P@Wh WMMA.
// ---------------------------------------------------------------------------
__global__ __launch_bounds__(128) void k_scores(const float* __restrict__ Wh,
                                                const float* __restrict__ a,
                                                float* __restrict__ s_src,
                                                float* __restrict__ s_dst,
                                                uint16_t* __restrict__ whb) {
  const int node = blockIdx.x;
  const int f    = threadIdx.x;
  const float v  = Wh[(size_t)node * F_DIM + f];
  whb[(size_t)node * F_DIM + f] = (uint16_t)f2bf_bits(v);
  float p1 = v * a[f];
  float p2 = v * a[F_DIM + f];
  #pragma unroll
  for (int off = 16; off > 0; off >>= 1) {
    p1 += __shfl_down(p1, off, 32);
    p2 += __shfl_down(p2, off, 32);
  }
  __shared__ float r1[4], r2[4];
  if ((f & 31) == 0) { r1[f >> 5] = p1; r2[f >> 5] = p2; }
  __syncthreads();
  if (f == 0) {
    s_src[node] = r1[0] + r1[1] + r1[2] + r1[3];
    s_dst[node] = r2[0] + r2[1] + r2[2] + r2[3];
  }
}

// ---------------------------------------------------------------------------
// Kernel 4: fused masked-softmax attention + aggregation (flash style).
// Grid: B * N/16 blocks of 288 threads (9 waves).
//   wave 0   : producer -- builds bf16 P-tile (16x32) in WMMA A layout + alpha
//   waves 1-8: consumers -- each owns a 16-wide F block, C-rescale + WMMA
// Block-start staging of s_dst + adj bitmask rows goes through the TDM
// (tensor_load_to_lds, incl. LDS pad feature for the 65-word row stride).
// ---------------------------------------------------------------------------
__global__ __launch_bounds__(288) void k_attn(const float* __restrict__ s_src,
                                              const float* __restrict__ s_dst,
                                              const unsigned* __restrict__ adjbits,
                                              const uint16_t* __restrict__ whb,
                                              float* __restrict__ out) {
  __shared__ __align__(16) float    lds_sdst[N_DIM];
  __shared__ __align__(16) unsigned lds_adjw[16 * 65];      // padded vs bank conflicts
  __shared__ __align__(32) unsigned lds_A[2][32][8];        // double-buffered A tile
  __shared__ __align__(16) float    lds_alpha[2][16];
  __shared__ __align__(16) float    lds_linv[16];

  const int blk  = blockIdx.x;
  const int b    = blk >> 7;
  const int i0   = (blk & 127) << 4;
  const int t    = threadIdx.x;
  const int lane = t & 31;
  const int wave = t >> 5;
  const int M    = lane & 15;
  const int half = lane >> 4;

#ifdef HAVE_TDM
  if (wave == 0) {
    // 1D copy: 2048 f32 s_dst row -> LDS.
    tdm_load_4B((unsigned)(size_t)&lds_sdst[0], s_dst + (size_t)b * N_DIM,
                N_DIM, 1, N_DIM, 0, N_DIM, 0, 0);
    // 2D copy: 16 rows x 64 words of adj bits, LDS pad 1 dword every 64
    // (pad_interval code 5) -> 65-word row stride in LDS.
    tdm_load_4B((unsigned)(size_t)&lds_adjw[0], adjbits + (size_t)i0 * 64,
                64, 16, 64, 16, 64, 1, 5);
    __builtin_amdgcn_s_wait_tensorcnt(0);
  }
#else
  for (int idx = t; idx < N_DIM / 4; idx += 288)
    ((float4*)lds_sdst)[idx] = ((const float4*)(s_dst + (size_t)b * N_DIM))[idx];
  for (int idx = t; idx < 16 * 64; idx += 288) {
    const int r = idx >> 6, c = idx & 63;
    lds_adjw[r * 65 + c] = adjbits[(size_t)(i0 + r) * 64 + c];
  }
#endif
  __syncthreads();

  float s_i = 0.f, mrun = -__builtin_inff(), lrun = 0.f;
  if (wave == 0) s_i = s_src[(size_t)b * N_DIM + i0 + M];

  v8f acc = {};
  const uint16_t* bbase = whb + (size_t)b * N_DIM * F_DIM + (wave - 1) * 16;

  for (int step = 0; step < N_DIM / 32; ++step) {
    const int j0  = step * 32;
    const int buf = step & 1;
    v16bf Bm{};
    if (wave == 0) {
      const unsigned word = lds_adjw[M * 65 + (j0 >> 5)];
      float p[16];
      float mx = -__builtin_inff();
      #pragma unroll
      for (int e = 0; e < 16; ++e) {
        // WMMA A (16x32 bf16) element->K map: K = (e/8)*16 + half*8 + (e%8)
        const int kk = ((e >> 3) << 4) + (half << 3) + (e & 7);
        float sc = s_i + lds_sdst[j0 + kk];
        sc = sc > 0.f ? sc : sc * SLOPE;
        sc = ((word >> kk) & 1u) ? sc : NEG_BIG_F;
        p[e] = sc;
        mx = fmaxf(mx, sc);
      }
      mx = fmaxf(mx, __shfl_xor(mx, 16, 32));        // halves cover disjoint K
      const float mnew  = fmaxf(mrun, mx);
      const float alpha = __expf(mrun - mnew);       // exp(-inf)=0 first step
      float sum = 0.f;
      unsigned packed[8];
      #pragma unroll
      for (int e = 0; e < 16; e += 2) {
        const float p0 = __expf(p[e]     - mnew);
        const float p1 = __expf(p[e + 1] - mnew);
        sum += p0 + p1;
        packed[e >> 1] = f2bf_bits(p0) | (f2bf_bits(p1) << 16);
      }
      sum += __shfl_xor(sum, 16, 32);
      lrun = lrun * alpha + sum;
      mrun = mnew;
      #pragma unroll
      for (int e = 0; e < 8; ++e) lds_A[buf][lane][e] = packed[e];
      if (half == 0) lds_alpha[buf][M] = alpha;
    } else {
      // B 32x16 bf16: lane%16 = N, element e -> K = (e/8)*16 + half*8 + (e%8).
      const uint16_t* bp = bbase + (size_t)j0 * F_DIM + M;
      #pragma unroll
      for (int e = 0; e < 16; ++e) {
        const int kk = ((e >> 3) << 4) + (half << 3) + (e & 7);
        Bm[e] = __builtin_bit_cast(__bf16, bp[(size_t)kk * F_DIM]);
      }
      if (step + 1 < N_DIM / 32)
        __builtin_prefetch(bbase + (size_t)(j0 + 32 + lane) * F_DIM, 0, 0);
    }
    __syncthreads();   // one barrier/iter; double buffer covers WAR on lds_A
    if (wave != 0) {
      const v16bf  Am  = __builtin_bit_cast(v16bf, *(const v8u*)&lds_A[buf][lane][0]);
      const float4 alo = *(const float4*)&lds_alpha[buf][half << 3];
      const float4 ahi = *(const float4*)&lds_alpha[buf][(half << 3) + 4];
      acc[0] *= alo.x; acc[1] *= alo.y; acc[2] *= alo.z; acc[3] *= alo.w;
      acc[4] *= ahi.x; acc[5] *= ahi.y; acc[6] *= ahi.z; acc[7] *= ahi.w;
      acc = __builtin_amdgcn_wmma_f32_16x16x32_bf16(false, Am, false, Bm,
                                                    (short)0, acc, false, false);
    }
  }

  if (wave == 0 && half == 0) lds_linv[M] = 1.0f / lrun;
  __syncthreads();
  if (wave != 0) {
    const float4 llo = *(const float4*)&lds_linv[half << 3];
    const float4 lhi = *(const float4*)&lds_linv[(half << 3) + 4];
    const float li[8] = {llo.x, llo.y, llo.z, llo.w, lhi.x, lhi.y, lhi.z, lhi.w};
    float* op = out + ((size_t)b * N_DIM + i0) * F_DIM + (wave - 1) * 16 + M;
    #pragma unroll
    for (int r = 0; r < 8; ++r)
      op[(size_t)(r + (half << 3)) * F_DIM] = acc[r] * li[r];
  }
}

// ---------------------------------------------------------------------------
extern "C" void kernel_launch(void* const* d_in, const int* in_sizes, int n_in,
                              void* d_out, int out_size, void* d_ws, size_t ws_size,
                              hipStream_t stream) {
  (void)in_sizes; (void)n_in; (void)out_size; (void)ws_size;
  const float* h   = (const float*)d_in[0];   // (8,2048,128) f32
  const int*   adj = (const int*)d_in[1];     // (2048,2048) i32
  const float* W   = (const float*)d_in[2];   // (128,128) f32
  const float* a   = (const float*)d_in[3];   // (256,) f32
  float*       out = (float*)d_out;           // (8,2048,128) f32

  char* ws = (char*)d_ws;
  float*    Wh      = (float*)ws;                               // 8 MB
  uint16_t* whb     = (uint16_t*)(ws + (size_t)(8  << 20));     // 4 MB
  float*    ssrc    = (float*)(ws + (size_t)(12 << 20));        // 64 KB
  float*    sdst    = (float*)(ws + (size_t)(12 << 20) + (64 << 10));
  unsigned* adjbits = (unsigned*)(ws + (size_t)(12 << 20) + (128 << 10)); // 512 KB

  k_pack  <<<(N_DIM * 64) / 256, 256, 0, stream>>>(adj, adjbits);
  k_wh    <<<B_DIM * (N_DIM / 16), 256, 0, stream>>>(h, W, Wh);
  k_scores<<<B_DIM * N_DIM,        128, 0, stream>>>(Wh, a, ssrc, sdst, whb);
  k_attn  <<<B_DIM * (N_DIM / 16), 288, 0, stream>>>(ssrc, sdst, adjbits, whb, out);
}
